// Pooling_3D_Layer_10084583211474
// MI455X (gfx1250) — matrix operand order repaired and checked by reading, not compiled, not run on hardware
//
#include <hip/hip_runtime.h>

typedef __bf16 bf16_t;
typedef __attribute__((ext_vector_type(16))) __bf16 v16bf;
typedef __attribute__((ext_vector_type(8)))  __bf16 v8bf;
typedef __attribute__((ext_vector_type(8)))  float  v8f;

#define N_FINE   8000
#define N_TOT    10000
#define DH       64
#define EDGE_K   192
#define NODE_K   128
#define VT_LD    10016
#define NEG_SLOPE 0.01f
#define CDIV(a,b) (((a)+(b)-1)/(b))

__device__ __forceinline__ float lrelu_f(float x) { return x > 0.f ? x : NEG_SLOPE * x; }

__device__ __forceinline__ v8f wmma_bf16(v16bf a, v16bf b, v8f c) {
  return __builtin_amdgcn_wmma_f32_16x16x32_bf16(false, a, false, b, (short)0, c,
                                                 false, false);
}

// A fragment (16x32 bf16) from row-major A[M][lda]
__device__ __forceinline__ v16bf load_a_frag(const bf16_t* __restrict__ A, int lda,
                                             int m0, int kc, int lane) {
  const bf16_t* row = A + (size_t)(m0 + (lane & 15)) * (size_t)lda;
  int kb = kc + ((lane & 16) ? 8 : 0);
  v8bf lo = *(const v8bf*)(row + kb);
  v8bf hi = *(const v8bf*)(row + kb + 16);
  return __builtin_shufflevector(lo, hi, 0,1,2,3,4,5,6,7,8,9,10,11,12,13,14,15);
}

// B fragment (32x16 bf16) from transposed storage Bt[n][ldb]
__device__ __forceinline__ v16bf load_bt_frag(const bf16_t* __restrict__ Bt, int ldb,
                                              int n0, int kc, int lane) {
  const bf16_t* row = Bt + (size_t)(n0 + (lane & 15)) * (size_t)ldb;
  int kb = kc + ((lane & 16) ? 8 : 0);
  v8bf lo = *(const v8bf*)(row + kb);
  v8bf hi = *(const v8bf*)(row + kb + 16);
  return __builtin_shufflevector(lo, hi, 0,1,2,3,4,5,6,7,8,9,10,11,12,13,14,15);
}

// C[M][64] = A[M][K] @ Bt^T ; one wave per 16x64 tile, 4 waves/block
__global__ void gemm_bf16_kernel(const bf16_t* __restrict__ A,
                                 const bf16_t* __restrict__ Bt,
                                 float* __restrict__ C, int M, int K) {
  int lane = threadIdx.x & 31;
  int mt = blockIdx.x * (blockDim.x >> 5) + (threadIdx.x >> 5);
  int m0 = mt * 16;
  if (m0 >= M) return;
  v8f a0 = {}, a1 = {}, a2 = {}, a3 = {};
  for (int kc = 0; kc < K; kc += 32) {
    v16bf a = load_a_frag(A, K, m0, kc, lane);
    a0 = wmma_bf16(a, load_bt_frag(Bt, K, 0,  kc, lane), a0);
    a1 = wmma_bf16(a, load_bt_frag(Bt, K, 16, kc, lane), a1);
    a2 = wmma_bf16(a, load_bt_frag(Bt, K, 32, kc, lane), a2);
    a3 = wmma_bf16(a, load_bt_frag(Bt, K, 48, kc, lane), a3);
  }
  int col = lane & 15;
  int r0 = m0 + ((lane & 16) ? 8 : 0);
#pragma unroll
  for (int r = 0; r < 8; ++r) {
    size_t ro = (size_t)(r0 + r) * DH;
    C[ro + col]      = a0[r];
    C[ro + 16 + col] = a1[r];
    C[ro + 32 + col] = a2[r];
    C[ro + 48 + col] = a3[r];
  }
}

// flash attention: computes S^T = K@Q^T so C-layout(S^T) == A-layout(P)
__global__ void attn_kernel(const bf16_t* __restrict__ Q,
                            const bf16_t* __restrict__ Km,
                            const bf16_t* __restrict__ Vt,
                            float* __restrict__ Cross) {
  int lane = threadIdx.x & 31;
  int qt = blockIdx.x * (blockDim.x >> 5) + (threadIdx.x >> 5);
  int q0 = qt * 16;
  if (q0 >= N_TOT) return;
  v16bf bq0 = load_bt_frag(Q, DH, q0, 0, lane);
  v16bf bq1 = load_bt_frag(Q, DH, q0, 32, lane);
  v8f o0 = {}, o1 = {}, o2 = {}, o3 = {};
  float mrun = -3.0e38f, lrun = 0.0f;
  int half8 = (lane & 16) ? 8 : 0;

  for (int j0 = 0; j0 < N_TOT; j0 += 32) {
    bool full = (j0 + 32 <= N_TOT);
    v8f s0 = {}, s1 = {};
    s0 = wmma_bf16(load_a_frag(Km, DH, j0, 0,  lane), bq0, s0);
    s0 = wmma_bf16(load_a_frag(Km, DH, j0, 32, lane), bq1, s0);
    if (full) {
      s1 = wmma_bf16(load_a_frag(Km, DH, j0 + 16, 0,  lane), bq0, s1);
      s1 = wmma_bf16(load_a_frag(Km, DH, j0 + 16, 32, lane), bq1, s1);
    }
    float cm = -3.0e38f;
#pragma unroll
    for (int r = 0; r < 8; ++r) cm = fmaxf(cm, s0[r]);
    if (full) {
#pragma unroll
      for (int r = 0; r < 8; ++r) cm = fmaxf(cm, s1[r]);
    }
    cm = fmaxf(cm, __shfl_xor(cm, 16, 32));
    float newm = fmaxf(mrun, cm);
    float alpha = __expf(mrun - newm);
    mrun = newm;
    float p0[8], p1[8];
    float rs = 0.f;
#pragma unroll
    for (int r = 0; r < 8; ++r) { p0[r] = __expf(s0[r] - mrun); rs += p0[r]; }
#pragma unroll
    for (int r = 0; r < 8; ++r) { p1[r] = full ? __expf(s1[r] - mrun) : 0.f; rs += p1[r]; }
    rs += __shfl_xor(rs, 16, 32);
    lrun = lrun * alpha + rs;
#pragma unroll
    for (int r = 0; r < 8; ++r) {
      float ar = __shfl(alpha, r + half8, 32);
      o0[r] *= ar; o1[r] *= ar; o2[r] *= ar; o3[r] *= ar;
    }
    v16bf pa;
#pragma unroll
    for (int r = 0; r < 8; ++r) { pa[r] = (bf16_t)p0[r]; pa[r + 8] = (bf16_t)p1[r]; }
    o0 = wmma_bf16(pa, load_bt_frag(Vt, VT_LD, 0,  j0, lane), o0);
    o1 = wmma_bf16(pa, load_bt_frag(Vt, VT_LD, 16, j0, lane), o1);
    o2 = wmma_bf16(pa, load_bt_frag(Vt, VT_LD, 32, j0, lane), o2);
    o3 = wmma_bf16(pa, load_bt_frag(Vt, VT_LD, 48, j0, lane), o3);
  }
  int col = lane & 15;
#pragma unroll
  for (int r = 0; r < 8; ++r) {
    float lr = __shfl(lrun, r + half8, 32);
    float inv = 1.0f / lr;
    size_t ro = (size_t)(q0 + r + half8) * DH;
    Cross[ro + col]      = o0[r] * inv;
    Cross[ro + 16 + col] = o1[r] * inv;
    Cross[ro + 32 + col] = o2[r] * inv;
    Cross[ro + 48 + col] = o3[r] * inv;
  }
}

__global__ void pack_wt_kernel(const float* __restrict__ W, bf16_t* __restrict__ Wt,
                               int K, int N, int Kpad) {
  int idx = blockIdx.x * blockDim.x + threadIdx.x;
  if (idx >= N * Kpad) return;
  int n = idx / Kpad, k = idx - n * Kpad;
  Wt[idx] = (bf16_t)((k < K) ? W[(size_t)k * N + n] : 0.0f);
}

__global__ void pack_bf16_kernel(const float* __restrict__ X, bf16_t* __restrict__ Y,
                                 long n, int do_lrelu) {
  long idx = (long)blockIdx.x * blockDim.x + threadIdx.x;
  if (idx >= n) return;
  float v = X[idx];
  if (do_lrelu) v = lrelu_f(v);
  Y[idx] = (bf16_t)v;
}

__global__ void packT_kernel(const float* __restrict__ X, bf16_t* __restrict__ Yt,
                             int rows, int cols, int ldt) {
  long idx = (long)blockIdx.x * blockDim.x + threadIdx.x;
  if (idx >= (long)cols * ldt) return;
  int c = (int)(idx / ldt);
  int r = (int)(idx - (long)c * ldt);
  Yt[idx] = (bf16_t)((r < rows) ? X[(size_t)r * cols + c] : 0.0f);
}

__global__ void zero_kernel(float* __restrict__ p, long n) {
  long idx = (long)blockIdx.x * blockDim.x + threadIdx.x;
  if (idx < n) p[idx] = 0.0f;
}

__global__ void edge_input_kernel(const float* __restrict__ fh, const float* __restrict__ ch,
                                  const float* __restrict__ fx, const float* __restrict__ px,
                                  const float* __restrict__ ef, const int* __restrict__ src,
                                  const int* __restrict__ dst, bf16_t* __restrict__ Xe, int E) {
  long idx = (long)blockIdx.x * blockDim.x + threadIdx.x;
  if (idx >= (long)E * EDGE_K) return;
  int e = (int)(idx / EDGE_K);
  int j = (int)(idx - (long)e * EDGE_K);
  int s = src[e], d = dst[e];
  float v = 0.f;
  if (j < 64)       v = fh[(size_t)s * DH + j];
  else if (j < 128) v = ch[(size_t)(d - N_FINE) * DH + (j - 64)];
  else if (j < 160) v = ef[(size_t)e * 32 + (j - 128)];
  else if (j < 175) {
    float dx = fx[s * 3 + 0] - px[d * 3 + 0];
    float dy = fx[s * 3 + 1] - px[d * 3 + 1];
    float dz = fx[s * 3 + 2] - px[d * 3 + 2];
    float d2 = dx * dx + dy * dy + dz * dz;
    float sig = powf(1.5f, (float)(j - 160));
    v = __expf(-d2 / sig);
  }
  Xe[idx] = (bf16_t)v;
}

__global__ void ln_kernel(const float* __restrict__ X, const float* __restrict__ bias,
                          const float* __restrict__ g, const float* __restrict__ be,
                          bf16_t* __restrict__ Yb, float* __restrict__ Yf, int do_lrelu) {
  int row = blockIdx.x, lane = threadIdx.x;
  size_t base = (size_t)row * DH;
  float x0 = X[base + lane] + bias[lane];
  float x1 = X[base + lane + 32] + bias[lane + 32];
  float s = x0 + x1, q = x0 * x0 + x1 * x1;
#pragma unroll
  for (int m = 16; m >= 1; m >>= 1) { s += __shfl_xor(s, m, 32); q += __shfl_xor(q, m, 32); }
  float mu = s * (1.0f / DH);
  float rstd = rsqrtf(q * (1.0f / DH) - mu * mu + 1e-5f);
  float y0 = g[lane] * (x0 - mu) * rstd + be[lane];
  float y1 = g[lane + 32] * (x1 - mu) * rstd + be[lane + 32];
  if (do_lrelu) { y0 = lrelu_f(y0); y1 = lrelu_f(y1); }
  if (Yf) { Yf[base + lane] = y0; Yf[base + lane + 32] = y1; }
  Yb[base + lane] = (bf16_t)y0;
  Yb[base + lane + 32] = (bf16_t)y1;
}

__global__ void coef_kernel(const float* __restrict__ H, const float* __restrict__ b1,
                            const float* __restrict__ w2, const float* __restrict__ b2,
                            float* __restrict__ coef) {
  int e = blockIdx.x, lane = threadIdx.x;
  size_t base = (size_t)e * DH;
  float x0 = lrelu_f(H[base + lane] + b1[lane]) * w2[lane];
  float x1 = lrelu_f(H[base + lane + 32] + b1[lane + 32]) * w2[lane + 32];
  float s = x0 + x1;
#pragma unroll
  for (int m = 16; m >= 1; m >>= 1) s += __shfl_xor(s, m, 32);
  if (lane == 0) coef[e] = s + b2[0];
}

__global__ void scatter_kernel(const float* __restrict__ msg, const float* __restrict__ coef,
                               const float* __restrict__ px, const int* __restrict__ src,
                               const int* __restrict__ dst, float* __restrict__ aggr,
                               float* __restrict__ cnt, float* __restrict__ xupd, int E) {
  long idx = (long)blockIdx.x * blockDim.x + threadIdx.x;
  if (idx >= (long)E * 16) return;
  int e = (int)(idx >> 4), gq = (int)(idx & 15);
  int d = dst[e];
  size_t mb = (size_t)e * DH + gq * 4;
  float* ab = aggr + (size_t)d * DH + gq * 4;
  atomicAdd(ab + 0, msg[mb + 0]);
  atomicAdd(ab + 1, msg[mb + 1]);
  atomicAdd(ab + 2, msg[mb + 2]);
  atomicAdd(ab + 3, msg[mb + 3]);
  if (gq == 0) {
    atomicAdd(cnt + d, 1.0f);
    int sn = src[e];
    float c = coef[e];
#pragma unroll
    for (int i = 0; i < 3; ++i)
      atomicAdd(xupd + (size_t)d * 3 + i, (px[sn * 3 + i] - px[d * 3 + i]) * c);
  }
}

__global__ void finalize_kernel(const float* __restrict__ cnt, const float* __restrict__ og,
                                const float* __restrict__ px, const float* __restrict__ xupd,
                                float* __restrict__ aggr, float* __restrict__ xev) {
  int idx = blockIdx.x * blockDim.x + threadIdx.x;
  if (idx >= N_TOT * DH) return;
  int node = idx >> 6, d = idx & 63;
  float den = fmaxf(cnt[node], 1.0f);
  aggr[idx] = aggr[idx] / den;
  if (d < 3)
    xev[node * 3 + d] = 0.25f * og[node * 3 + d] + 0.75f * px[node * 3 + d]
                      + xupd[node * 3 + d] / den;
}

__global__ void node_concat_kernel(const float* __restrict__ ph, const float* __restrict__ cross,
                                   const float* __restrict__ aggr, float* __restrict__ hbuf,
                                   bf16_t* __restrict__ cat) {
  int idx = blockIdx.x * blockDim.x + threadIdx.x;
  if (idx >= N_TOT * DH) return;
  int node = idx >> 6, d = idx & 63;
  float h = ph[idx] + cross[idx];
  hbuf[idx] = h;
  cat[(size_t)node * NODE_K + d] = (bf16_t)h;
  cat[(size_t)node * NODE_K + DH + d] = (bf16_t)aggr[idx];
}

__global__ void final_ln_kernel(const float* __restrict__ X, const float* __restrict__ bias,
                                const float* __restrict__ g, const float* __restrict__ be,
                                const float* __restrict__ h, float* __restrict__ out) {
  int row = blockIdx.x, lane = threadIdx.x;
  size_t base = (size_t)row * DH;
  float x0 = X[base + lane] + bias[lane];
  float x1 = X[base + lane + 32] + bias[lane + 32];
  float s = x0 + x1, q = x0 * x0 + x1 * x1;
#pragma unroll
  for (int m = 16; m >= 1; m >>= 1) { s += __shfl_xor(s, m, 32); q += __shfl_xor(q, m, 32); }
  float mu = s * (1.0f / DH);
  float rstd = rsqrtf(q * (1.0f / DH) - mu * mu + 1e-5f);
  float y0 = g[lane] * (x0 - mu) * rstd + be[lane];
  float y1 = g[lane + 32] * (x1 - mu) * rstd + be[lane + 32];
  out[base + lane]      = 0.5f * y0 + 0.5f * h[base + lane];
  out[base + lane + 32] = 0.5f * y1 + 0.5f * h[base + lane + 32];
}

extern "C" void kernel_launch(void* const* d_in, const int* in_sizes, int n_in,
                              void* d_out, int out_size, void* d_ws, size_t ws_size,
                              hipStream_t stream) {
  (void)n_in; (void)out_size; (void)ws_size;
  const int E = in_sizes[14];

  const float* P[23];
  {
    static const int canon2alpha[23] = {7,9,13,11,8,10,14,12,1,0,2,
                                        15,17,21,19,16,18,22,20,3,5,4,6};
    bool insertion = (in_sizes[18] == 175 * 64);
    for (int i = 0; i < 23; ++i)
      P[i] = (const float*)d_in[18 + (insertion ? i : canon2alpha[i])];
  }

  char* ws = (char*)d_ws;
  size_t off = 0;
  auto alloc = [&](size_t bytes) -> void* {
    void* p = ws + off;
    off = (off + bytes + 255) & ~(size_t)255;
    return p;
  };

  bf16_t* eW1t = (bf16_t*)alloc((size_t)DH * EDGE_K * 2);
  bf16_t* eW2t = (bf16_t*)alloc((size_t)DH * DH * 2);
  bf16_t* cW1t = (bf16_t*)alloc((size_t)DH * DH * 2);
  bf16_t* wqt  = (bf16_t*)alloc((size_t)DH * DH * 2);
  bf16_t* wkt  = (bf16_t*)alloc((size_t)DH * DH * 2);
  bf16_t* wvt  = (bf16_t*)alloc((size_t)DH * DH * 2);
  bf16_t* nW1t = (bf16_t*)alloc((size_t)DH * NODE_K * 2);
  bf16_t* nW2t = (bf16_t*)alloc((size_t)DH * DH * 2);
  bf16_t* Xe    = (bf16_t*)alloc((size_t)E * EDGE_K * 2);
  float*  e_t   = (float*) alloc((size_t)E * DH * 4);
  bf16_t* e_b   = (bf16_t*)alloc((size_t)E * DH * 2);
  float*  msg_f = (float*) alloc((size_t)E * DH * 4);
  bf16_t* msg_b = (bf16_t*)alloc((size_t)E * DH * 2);
  float*  coef  = (float*) alloc((size_t)E * 4);
  float*  cnt   = (float*) alloc((size_t)N_TOT * 4 * 4);
  float*  xupd  = cnt + N_TOT;
  float*  q_f   = (float*) alloc((size_t)N_TOT * DH * 4);
  bf16_t* ph_b  = (bf16_t*)alloc((size_t)N_TOT * DH * 2);
  float*  cross = (float*) alloc((size_t)N_TOT * DH * 4);
  float*  hbuf  = (float*) alloc((size_t)N_TOT * DH * 4);
  bf16_t* cat   = (bf16_t*)alloc((size_t)N_TOT * NODE_K * 2);
  float*  n_t   = (float*) alloc((size_t)N_TOT * DH * 4);
  bf16_t* n_b   = (bf16_t*)alloc((size_t)N_TOT * DH * 2);
  float*  aggrS[2]; bf16_t* QbS[2]; bf16_t* KbS[2]; bf16_t* VtS[2];
  for (int s = 0; s < 2; ++s) {
    aggrS[s] = (float*) alloc((size_t)N_TOT * DH * 4);
    QbS[s]   = (bf16_t*)alloc((size_t)N_TOT * DH * 2);
    KbS[s]   = (bf16_t*)alloc((size_t)N_TOT * DH * 2);
    VtS[s]   = (bf16_t*)alloc((size_t)DH * VT_LD * 2);
  }

  float* out = (float*)d_out;
  const long upd_off[2] = {0, 670000};
  const long xev_off[2] = {640000, 1310000};

  pack_wt_kernel<<<CDIV(DH * EDGE_K, 256), 256, 0, stream>>>(P[0],  eW1t, 175, DH, EDGE_K);
  pack_wt_kernel<<<CDIV(DH * DH, 256),     256, 0, stream>>>(P[4],  eW2t, DH, DH, DH);
  pack_wt_kernel<<<CDIV(DH * DH, 256),     256, 0, stream>>>(P[19], cW1t, DH, DH, DH);
  pack_wt_kernel<<<CDIV(DH * DH, 256),     256, 0, stream>>>(P[8],  wqt,  DH, DH, DH);
  pack_wt_kernel<<<CDIV(DH * DH, 256),     256, 0, stream>>>(P[9],  wkt,  DH, DH, DH);
  pack_wt_kernel<<<CDIV(DH * DH, 256),     256, 0, stream>>>(P[10], wvt,  DH, DH, DH);
  pack_wt_kernel<<<CDIV(DH * NODE_K, 256), 256, 0, stream>>>(P[11], nW1t, NODE_K, DH, NODE_K);
  pack_wt_kernel<<<CDIV(DH * DH, 256),     256, 0, stream>>>(P[15], nW2t, DH, DH, DH);

  const int gemmBlocksE = CDIV(E / 16, 4);
  const int gemmBlocksN = CDIV(N_TOT / 16, 4);

  for (int s = 0; s < 2; ++s) {
    const float* fh = (const float*)d_in[0 + s];
    const float* ch = (const float*)d_in[2 + s];
    const float* fx = (const float*)d_in[4 + s];
    const float* px = (const float*)d_in[8 + s];
    const float* og = (const float*)d_in[10 + s];
    const float* ef = (const float*)d_in[12 + s];
    const int* esrc = (const int*)d_in[14 + 2 * s];
    const int* edst = (const int*)d_in[15 + 2 * s];

    edge_input_kernel<<<(int)CDIV((long)E * EDGE_K, 256), 256, 0, stream>>>(
        fh, ch, fx, px, ef, esrc, edst, Xe, E);
    gemm_bf16_kernel<<<gemmBlocksE, 128, 0, stream>>>(Xe, eW1t, e_t, E, EDGE_K);
    ln_kernel<<<E, 32, 0, stream>>>(e_t, P[1], P[2], P[3], e_b, nullptr, 1);
    gemm_bf16_kernel<<<gemmBlocksE, 128, 0, stream>>>(e_b, eW2t, e_t, E, DH);
    ln_kernel<<<E, 32, 0, stream>>>(e_t, P[5], P[6], P[7], msg_b, msg_f, 0);
    gemm_bf16_kernel<<<gemmBlocksE, 128, 0, stream>>>(msg_b, cW1t, e_t, E, DH);
    coef_kernel<<<E, 32, 0, stream>>>(e_t, P[20], P[21], P[22], coef);

    zero_kernel<<<CDIV(N_TOT * DH, 256), 256, 0, stream>>>(aggrS[s], (long)N_TOT * DH);
    zero_kernel<<<CDIV(N_TOT * 4, 256), 256, 0, stream>>>(cnt, (long)N_TOT * 4);
    scatter_kernel<<<(int)CDIV((long)E * 16, 256), 256, 0, stream>>>(
        msg_f, coef, px, esrc, edst, aggrS[s], cnt, xupd, E);
    finalize_kernel<<<CDIV(N_TOT * DH, 256), 256, 0, stream>>>(
        cnt, og, px, xupd, aggrS[s], out + xev_off[s]);
  }

  for (int s = 0; s < 2; ++s) {
    const float* ph = (const float*)d_in[6 + s];
    pack_bf16_kernel<<<CDIV(N_TOT * DH, 256), 256, 0, stream>>>(ph, ph_b, (long)N_TOT * DH, 0);
    gemm_bf16_kernel<<<gemmBlocksN, 128, 0, stream>>>(ph_b, wqt, q_f, N_TOT, DH);
    pack_bf16_kernel<<<CDIV(N_TOT * DH, 256), 256, 0, stream>>>(q_f, QbS[s], (long)N_TOT * DH, 1);
    gemm_bf16_kernel<<<gemmBlocksN, 128, 0, stream>>>(ph_b, wkt, q_f, N_TOT, DH);
    pack_bf16_kernel<<<CDIV(N_TOT * DH, 256), 256, 0, stream>>>(q_f, KbS[s], (long)N_TOT * DH, 1);
    gemm_bf16_kernel<<<gemmBlocksN, 128, 0, stream>>>(ph_b, wvt, q_f, N_TOT, DH);
    packT_kernel<<<CDIV(DH * VT_LD, 256), 256, 0, stream>>>(q_f, VtS[s], N_TOT, DH, VT_LD);
  }

  for (int s = 0; s < 2; ++s) {
    const float* ph = (const float*)d_in[6 + s];
    attn_kernel<<<gemmBlocksN, 128, 0, stream>>>(QbS[s], KbS[1 - s], VtS[1 - s], cross);
    node_concat_kernel<<<CDIV(N_TOT * DH, 256), 256, 0, stream>>>(ph, cross, aggrS[s], hbuf, cat);
    gemm_bf16_kernel<<<gemmBlocksN, 128, 0, stream>>>(cat, nW1t, n_t, N_TOT, NODE_K);
    ln_kernel<<<N_TOT, 32, 0, stream>>>(n_t, P[12], P[13], P[14], n_b, nullptr, 1);
    gemm_bf16_kernel<<<gemmBlocksN, 128, 0, stream>>>(n_b, nW2t, n_t, N_TOT, DH);
    final_ln_kernel<<<N_TOT, 32, 0, stream>>>(n_t, P[16], P[17], P[18], hbuf, out + upd_off[s]);
  }
}
